// SemanticSelector_47090021433782
// MI455X (gfx1250) — matrix operand order repaired and checked by reading, not compiled, not run on hardware
//
#include <hip/hip_runtime.h>
#include <hip/hip_bf16.h>

#define BDIM 16384
#define EDIM 1024
#define RDIM 2048
#define TOPKK 80

#define BM 128
#define BN 128
#define BK 32
#define LDT 40   // LDS row pitch in bf16 elements (80B: 16B-aligned rows, conflict-free b128 reads)

typedef __attribute__((ext_vector_type(16))) __bf16 v16bf;
typedef __attribute__((ext_vector_type(8)))  __bf16 v8bf;
typedef __attribute__((ext_vector_type(8)))  float  v8f;
typedef __attribute__((ext_vector_type(4)))  int    v4i_;

// --- CDNA5 async global->LDS path (guarded: falls back to sync staging) ---
#if defined(__has_builtin)
#  if __has_builtin(__builtin_amdgcn_global_load_async_to_lds_b128)
#    define HAS_ASYNC_LDS 1
#  endif
#endif
#ifndef HAS_ASYNC_LDS
#  define HAS_ASYNC_LDS 0
#endif

#if HAS_ASYNC_LDS
typedef __attribute__((address_space(1))) v4i_ gas_v4i;
typedef __attribute__((address_space(3))) v4i_ las_v4i;
__device__ __forceinline__ void async_copy16(const void* g, void* l) {
    __builtin_amdgcn_global_load_async_to_lds_b128((gas_v4i*)g, (las_v4i*)l, 0, 0);
}
#if __has_builtin(__builtin_amdgcn_s_wait_asynccnt)
#  define WAIT_ASYNC(n) __builtin_amdgcn_s_wait_asynccnt(n)
#else
#  define WAIT_ASYNC(n) asm volatile("s_wait_asynccnt %0" ::"i"(n) : "memory")
#endif
#endif

__device__ __forceinline__ float sigf(float x) { return 1.0f / (1.0f + __expf(-x)); }

// ---------------------------------------------------------------------------
// f32 -> bf16 conversion (grid-stride)
// ---------------------------------------------------------------------------
__global__ void f32_to_bf16_kernel(const float* __restrict__ src, __bf16* __restrict__ dst, int n)
{
    int i = blockIdx.x * blockDim.x + threadIdx.x;
    int stride = gridDim.x * blockDim.x;
    for (; i < n; i += stride) dst[i] = (__bf16)src[i];
}

// ---------------------------------------------------------------------------
// GEMM: out[M,N] = epilogue( A[M,K] @ W[N,K]^T + bias[N] )
// M = BDIM, N = K = EDIM, bf16 in, f32 accumulate via v_wmma_f32_16x16x32_bf16.
// 256 threads = 8 waves (4 M x 2 N); each wave: 32x64 tile = 2x4 16x16 frags.
// Double-buffered LDS tiles filled by GLOBAL_LOAD_ASYNC_TO_LDS_B128 (ASYNCcnt);
// steady-state loop is branch-free (last stage peeled), so each iteration:
//   issue next stage's 4 async DMAs -> s_wait_asynccnt 4 -> barrier -> 8 WMMA -> barrier.
// EPI: 0 = relu->bf16, 1 = linear->bf16, 2 = sigmoid(aux)*x->bf16, 3 = linear->f32
// ---------------------------------------------------------------------------
template <int EPI>
__global__ __launch_bounds__(256)
void gemm_epi(const __bf16* __restrict__ A, const __bf16* __restrict__ W,
              const float* __restrict__ bias, const __bf16* __restrict__ aux,
              __bf16* __restrict__ outb, float* __restrict__ outf)
{
#if HAS_ASYNC_LDS
    __shared__ __bf16 lA[2][BM * LDT];
    __shared__ __bf16 lW[2][BN * LDT];
#else
    __shared__ __bf16 lA[1][BM * LDT];
    __shared__ __bf16 lW[1][BN * LDT];
#endif

    const int tid   = threadIdx.x;
    const int lane  = tid & 31;
    const int wave  = tid >> 5;
    const int waveM = wave & 3;   // 0..3
    const int waveN = wave >> 2;  // 0..1
    const int tileM = blockIdx.x * BM;
    const int tileN = blockIdx.y * BN;

    // global->LDS staging: 2 threads per row, 16 bf16 (32B) each
    const int ldrow = tid >> 1;        // 0..127
    const int ldcol = (tid & 1) * 16;  // 0 or 16

    v8f acc[2][4];
#pragma unroll
    for (int i = 0; i < 2; ++i)
#pragma unroll
        for (int j = 0; j < 4; ++j)
#pragma unroll
            for (int k = 0; k < 8; ++k) acc[i][j][k] = 0.0f;

    const __bf16* aRow = A + (size_t)(tileM + ldrow) * EDIM + ldcol;
    const __bf16* wRow = W + (size_t)(tileN + ldrow) * EDIM + ldcol;
    const int lOff = ldrow * LDT + ldcol;

    const int m = lane & 15;
    const int h = lane >> 4;

    // Fragment-load + 8x WMMA for one staged K-slab.
    auto compute = [&](const __bf16* Abuf, const __bf16* Wbuf) {
        // A fragments: lane(m,h): elems 0..7 -> K = 8h..8h+7 ; elems 8..15 -> K = 16+8h..
        v16bf af[2];
#pragma unroll
        for (int mf = 0; mf < 2; ++mf) {
            const __bf16* p = Abuf + (waveM * 32 + mf * 16 + m) * LDT;
            v8bf c0 = *(const v8bf*)(p + 8 * h);
            v8bf c1 = *(const v8bf*)(p + 16 + 8 * h);
#pragma unroll
            for (int i = 0; i < 8; ++i) { af[mf][i] = c0[i]; af[mf][i + 8] = c1[i]; }
        }
        // B fragments: lane(n,h): elems 0..7 -> K = 8h..8h+7 ; elems 8..15 -> K = 16+8h.. (column n of B = row n of W)
        v16bf bfr[4];
#pragma unroll
        for (int nf = 0; nf < 4; ++nf) {
            const __bf16* p = Wbuf + (waveN * 64 + nf * 16 + m) * LDT;
            v8bf c0 = *(const v8bf*)(p + 8 * h);
            v8bf c1 = *(const v8bf*)(p + 16 + 8 * h);
#pragma unroll
            for (int i = 0; i < 8; ++i) { bfr[nf][i] = c0[i]; bfr[nf][i + 8] = c1[i]; }
        }
#pragma unroll
        for (int mf = 0; mf < 2; ++mf)
#pragma unroll
            for (int nf = 0; nf < 4; ++nf)
                acc[mf][nf] = __builtin_amdgcn_wmma_f32_16x16x32_bf16(
                    false, af[mf], false, bfr[nf], (short)0, acc[mf][nf], false, false);
    };

#if HAS_ASYNC_LDS
    auto stage = [&](int b, int kt) {
        async_copy16(aRow + kt,     &lA[b][lOff]);
        async_copy16(aRow + kt + 8, &lA[b][lOff + 8]);
        async_copy16(wRow + kt,     &lW[b][lOff]);
        async_copy16(wRow + kt + 8, &lW[b][lOff + 8]);
    };

    // prologue: stage kt=0 into buffer 0
    stage(0, 0);

    int buf = 0;
    for (int kt = 0; kt < EDIM - BK; kt += BK) {   // steady state: branch-free body
        stage(buf ^ 1, kt + BK);
        WAIT_ASYNC(4);   // current stage's 4 loads done; next stage in flight
        __syncthreads();
        compute(&lA[buf][0], &lW[buf][0]);
        __syncthreads();   // all waves done reading buf before next stage overwrites it
        buf ^= 1;
    }
    // peeled last stage
    WAIT_ASYNC(0);
    __syncthreads();
    compute(&lA[buf][0], &lW[buf][0]);
#else
    for (int kt = 0; kt < EDIM; kt += BK) {
        uint4 a0 = *(const uint4*)(aRow + kt);
        uint4 a1 = *(const uint4*)(aRow + kt + 8);
        uint4 w0 = *(const uint4*)(wRow + kt);
        uint4 w1 = *(const uint4*)(wRow + kt + 8);
        __syncthreads();
        *(uint4*)(&lA[0][lOff])     = a0;
        *(uint4*)(&lA[0][lOff + 8]) = a1;
        *(uint4*)(&lW[0][lOff])     = w0;
        *(uint4*)(&lW[0][lOff + 8]) = w1;
        __syncthreads();
        compute(&lA[0][0], &lW[0][0]);
    }
#endif

    // Epilogue. C/D layout: lane(n = lane&15, h): VGPR r -> (M = r + 8h, N = n)
    const int cn = lane & 15;
    const int ch = lane >> 4;
#pragma unroll
    for (int mf = 0; mf < 2; ++mf) {
#pragma unroll
        for (int nf = 0; nf < 4; ++nf) {
            const int col = tileN + waveN * 64 + nf * 16 + cn;
            const float bv = bias[col];
#pragma unroll
            for (int r = 0; r < 8; ++r) {
                const int row = tileM + waveM * 32 + mf * 16 + r + 8 * ch;
                const size_t idx = (size_t)row * EDIM + col;
                const float v = acc[mf][nf][r] + bv;
                if (EPI == 0) {
                    outb[idx] = (__bf16)fmaxf(v, 0.0f);
                } else if (EPI == 1) {
                    outb[idx] = (__bf16)v;
                } else if (EPI == 2) {
                    const float u = (float)aux[idx];
                    outb[idx] = (__bf16)(sigf(u) * v);
                } else {
                    outf[idx] = v;
                }
            }
        }
    }
}

// ---------------------------------------------------------------------------
// Per-row L2 norm (one 256-thread block per row)
// ---------------------------------------------------------------------------
__global__ __launch_bounds__(256)
void rownorm_kernel(const float* __restrict__ x, float* __restrict__ nrm)
{
    __shared__ float red[256];
    const int row = blockIdx.x;
    const float* p = x + (size_t)row * EDIM;
    float s = 0.0f;
    for (int c = threadIdx.x; c < EDIM; c += 256) { float v = p[c]; s += v * v; }
    red[threadIdx.x] = s;
    __syncthreads();
    for (int off = 128; off > 0; off >>= 1) {
        if (threadIdx.x < off) red[threadIdx.x] += red[threadIdx.x + off];
        __syncthreads();
    }
    if (threadIdx.x == 0) nrm[row] = sqrtf(red[0]);
}

// ---------------------------------------------------------------------------
// In-place: out1 = sigmoid(out1/||.||) * (out2/||.||) ; out2 = out2/||.||
// ---------------------------------------------------------------------------
__global__ void finalize_kernel(float* __restrict__ out1, float* __restrict__ out2,
                                const float* __restrict__ ng, const float* __restrict__ nl)
{
    int i = blockIdx.x * blockDim.x + threadIdx.x;
    const int stride = gridDim.x * blockDim.x;
    const int total = BDIM * EDIM;
    for (; i < total; i += stride) {
        const int row = i >> 10;
        const float f1 = out1[i] / fmaxf(ng[row], 1e-12f);
        const float f2 = out2[i] / fmaxf(nl[row], 1e-12f);
        out2[i] = f2;
        out1[i] = sigf(f1) * f2;
    }
}

// ---------------------------------------------------------------------------
// Top-80 by |v| per row (exact rank with lowest-index tie-break) + concat
// fused[row] = [ visual[row] (2048) | sparse(final[row]) (1024) ]
// ---------------------------------------------------------------------------
__global__ __launch_bounds__(256)
void topk_concat_kernel(const float* __restrict__ finalf, const float* __restrict__ visual,
                        float* __restrict__ fused)
{
    __shared__ float sab[EDIM];
    __shared__ float sva[EDIM];
    const int row = blockIdx.x;
    const float* fr = finalf + (size_t)row * EDIM;
    float* outr = fused + (size_t)row * (RDIM + EDIM);

    for (int c = threadIdx.x; c < EDIM; c += 256) {
        const float v = fr[c];
        sva[c] = v;
        sab[c] = fabsf(v);
    }
    __syncthreads();
    for (int c = threadIdx.x; c < EDIM; c += 256) {
        const float a = sab[c];
        int cnt = 0;
        for (int t = 0; t < EDIM; ++t) {
            const float b = sab[t];
            cnt += (b > a) || (b == a && t < c);
        }
        outr[RDIM + c] = (cnt < TOPKK) ? sva[c] : 0.0f;
    }
    for (int c = threadIdx.x; c < RDIM; c += 256)
        outr[c] = visual[(size_t)row * RDIM + c];
}

// ---------------------------------------------------------------------------
// Host orchestration
// ---------------------------------------------------------------------------
extern "C" void kernel_launch(void* const* d_in, const int* in_sizes, int n_in,
                              void* d_out, int out_size, void* d_ws, size_t ws_size,
                              hipStream_t stream)
{
    const float* sg_in  = (const float*)d_in[0];
    const float* sl_in  = (const float*)d_in[1];
    const float* visual = (const float*)d_in[2];
    const float* Wgu1 = (const float*)d_in[3];  const float* bgu1 = (const float*)d_in[4];
    const float* Wgu2 = (const float*)d_in[5];  const float* bgu2 = (const float*)d_in[6];
    const float* Wgd1 = (const float*)d_in[7];  const float* bgd1 = (const float*)d_in[8];
    const float* Wgd2 = (const float*)d_in[9];  const float* bgd2 = (const float*)d_in[10];
    const float* Wlu1 = (const float*)d_in[11]; const float* blu1 = (const float*)d_in[12];
    const float* Wlu2 = (const float*)d_in[13]; const float* blu2 = (const float*)d_in[14];
    const float* Wld1 = (const float*)d_in[15]; const float* bld1 = (const float*)d_in[16];
    const float* Wld2 = (const float*)d_in[17]; const float* bld2 = (const float*)d_in[18];
    const float* Wo   = (const float*)d_in[19]; const float* bo   = (const float*)d_in[20];
    const float* Win  = (const float*)d_in[21]; const float* bin_ = (const float*)d_in[22];

    float* out    = (float*)d_out;
    float* fused  = out;                                   // B x (R+E)
    float* finalb = out + (size_t)BDIM * (RDIM + EDIM);    // B x E : ag -> final_fused_feat
    float* fused2 = finalb + (size_t)BDIM * EDIM;          // B x E : al -> fused_feat2

    char* ws = (char*)d_ws;
    size_t o = 0;
    __bf16* XG = (__bf16*)(ws + o); o += (size_t)BDIM * EDIM * 2;   // semantic_global bf16; later sg
    __bf16* XL = (__bf16*)(ws + o); o += (size_t)BDIM * EDIM * 2;   // semantic_local  bf16; later sl
    __bf16* Wb[10];
    for (int i = 0; i < 10; ++i) { Wb[i] = (__bf16*)(ws + o); o += (size_t)EDIM * EDIM * 2; }
    __bf16* Hb = (__bf16*)(ws + o); o += (size_t)BDIM * EDIM * 2;   // hidden / value scratch
    __bf16* Ub = (__bf16*)(ws + o); o += (size_t)BDIM * EDIM * 2;   // gate (pre-sigmoid) scratch
    float*  ng = (float*)(ws + o);  o += (size_t)BDIM * 4;
    float*  nl = (float*)(ws + o);  o += (size_t)BDIM * 4;

    // --- convert inputs & weights to bf16 ---
    f32_to_bf16_kernel<<<4096, 256, 0, stream>>>(sg_in, XG, BDIM * EDIM);
    f32_to_bf16_kernel<<<4096, 256, 0, stream>>>(sl_in, XL, BDIM * EDIM);
    const float* wsrc[10] = { Wgu1, Wgu2, Wgd1, Wgd2, Wlu1, Wlu2, Wld1, Wld2, Wo,
                              Win + (size_t)2 * EDIM * EDIM };   // Wv = Win[2E:]
    for (int i = 0; i < 10; ++i)
        f32_to_bf16_kernel<<<1024, 256, 0, stream>>>(wsrc[i], Wb[i], EDIM * EDIM);
    const float* bv = bin_ + 2 * EDIM;

    const dim3 gg(BDIM / BM, EDIM / BN);   // 128 x 8
    const dim3 gb(256);

    // --- global path: sg = sigmoid(mlp_u(x)) * mlp_d(x) ---
    gemm_epi<0><<<gg, gb, 0, stream>>>(XG, Wb[0], bgu1, nullptr, Hb, nullptr); // h = relu(x Wgu1^T + b)
    gemm_epi<1><<<gg, gb, 0, stream>>>(Hb, Wb[1], bgu2, nullptr, Ub, nullptr); // u = h Wgu2^T + b
    gemm_epi<0><<<gg, gb, 0, stream>>>(XG, Wb[2], bgd1, nullptr, Hb, nullptr); // h = relu(x Wgd1^T + b)
    gemm_epi<2><<<gg, gb, 0, stream>>>(Hb, Wb[3], bgd2, Ub, XG, nullptr);      // sg = sig(u)*(h Wgd2^T + b)

    // --- local path: sl ---
    gemm_epi<0><<<gg, gb, 0, stream>>>(XL, Wb[4], blu1, nullptr, Hb, nullptr);
    gemm_epi<1><<<gg, gb, 0, stream>>>(Hb, Wb[5], blu2, nullptr, Ub, nullptr);
    gemm_epi<0><<<gg, gb, 0, stream>>>(XL, Wb[6], bld1, nullptr, Hb, nullptr);
    gemm_epi<2><<<gg, gb, 0, stream>>>(Hb, Wb[7], bld2, Ub, XL, nullptr);      // sl

    // --- attn_out for both paths: (s Wv^T + bv) Wo^T + bo ---
    gemm_epi<1><<<gg, gb, 0, stream>>>(XG, Wb[9], bv, nullptr, Hb, nullptr);   // vg
    gemm_epi<3><<<gg, gb, 0, stream>>>(Hb, Wb[8], bo, nullptr, nullptr, finalb); // ag (pre-norm)
    gemm_epi<1><<<gg, gb, 0, stream>>>(XL, Wb[9], bv, nullptr, Hb, nullptr);   // vl
    gemm_epi<3><<<gg, gb, 0, stream>>>(Hb, Wb[8], bo, nullptr, nullptr, fused2); // al (pre-norm)

    // --- l2norm + sigmoid-gate fuse ---
    rownorm_kernel<<<BDIM, gb, 0, stream>>>(finalb, ng);
    rownorm_kernel<<<BDIM, gb, 0, stream>>>(fused2, nl);
    finalize_kernel<<<4096, gb, 0, stream>>>(finalb, fused2, ng, nl);

    // --- top-80 sparsify + concat with visual ---
    topk_concat_kernel<<<BDIM, gb, 0, stream>>>(finalb, visual, fused);

    (void)in_sizes; (void)n_in; (void)out_size; (void)ws_size;
}
